// BasicTransformerBlock_23218593202526
// MI455X (gfx1250) — matrix operand order repaired and checked by reading, not compiled, not run on hardware
//
#include <hip/hip_runtime.h>
#include <cstdint>
#include <cstddef>

typedef _Float16 f16;
typedef __attribute__((ext_vector_type(16))) _Float16 v16h;
typedef __attribute__((ext_vector_type(8)))  float    v8f;

union HFrag { v16h v; unsigned int u[8]; };

#define TBS 256

#if __has_builtin(__builtin_amdgcn_global_load_async_to_lds_b128)
#define ASYNC_LDS 1
// GCC-style vector to exactly match the builtin's parameter pointee type.
typedef int v4i_b128 __attribute__((vector_size(16)));
typedef __attribute__((address_space(1))) v4i_b128* g_v4i_ptr;
typedef __attribute__((address_space(3))) v4i_b128* l_v4i_ptr;
#else
#define ASYNC_LDS 0
#endif

// 32-byte global -> LDS copy (two 16B beats); async on CDNA5 when available.
__device__ __forceinline__ void copy32_g2l(f16* lds_dst, const f16* gsrc) {
#if ASYNC_LDS
  g_v4i_ptr g = (g_v4i_ptr)(void*)const_cast<f16*>(gsrc);
  l_v4i_ptr l = (l_v4i_ptr)(void*)lds_dst;
  __builtin_amdgcn_global_load_async_to_lds_b128(g, l, 0, 0);
  __builtin_amdgcn_global_load_async_to_lds_b128(g, l, 16, 0);
#else
  ((uint4*)lds_dst)[0] = ((const uint4*)gsrc)[0];
  ((uint4*)lds_dst)[1] = ((const uint4*)gsrc)[1];
#endif
}

// 16-byte global -> LDS copy; async on CDNA5 when available.
__device__ __forceinline__ void copy16_g2l(f16* lds_dst, const f16* gsrc) {
#if ASYNC_LDS
  __builtin_amdgcn_global_load_async_to_lds_b128(
      (g_v4i_ptr)(void*)const_cast<f16*>(gsrc), (l_v4i_ptr)(void*)lds_dst, 0, 0);
#else
  *(uint4*)lds_dst = *(const uint4*)gsrc;
#endif
}

__device__ __forceinline__ void wait_async_lds() {
#if ASYNC_LDS
#if __has_builtin(__builtin_amdgcn_s_wait_asynccnt)
  __builtin_amdgcn_s_wait_asynccnt(0);
#else
  asm volatile("s_wait_asynccnt 0x0" ::: "memory");
#endif
#endif
}

// ---------------------------------------------------------------- reductions
__device__ __forceinline__ float block_reduce_sum(float v, float* red) {
#pragma unroll
  for (int o = 16; o > 0; o >>= 1) v += __shfl_xor(v, o, 32);
  const int tid = threadIdx.x;
  if ((tid & 31) == 0) red[tid >> 5] = v;
  __syncthreads();
  float t = 0.f;
#pragma unroll
  for (int i = 0; i < 8; ++i) t += red[i];
  __syncthreads();
  return t;
}

__device__ __forceinline__ float block_reduce_max(float v, float* red) {
#pragma unroll
  for (int o = 16; o > 0; o >>= 1) v = fmaxf(v, __shfl_xor(v, o, 32));
  const int tid = threadIdx.x;
  if ((tid & 31) == 0) red[tid >> 5] = v;
  __syncthreads();
  float t = -3.0e38f;
#pragma unroll
  for (int i = 0; i < 8; ++i) t = fmaxf(t, red[i]);
  __syncthreads();
  return t;
}

// ---------------------------------------------------------------- WMMA GEMM
// C[M,N] = A[M,K](f16) * B[K,N] (+bias +residual), out f32 and/or f16.
// BMODE 0: B is f32 row-major [K,N]   (weights; converted to f16 in staging)
// BMODE 1: B is f16 row-major [K,N]   (P·V)
// BMODE 2: B is f16, logical B[k][n] = Bsrc[n*ldb + k]  (Q·K^T, NT gemm)
template <int BMODE>
__global__ __launch_bounds__(TBS)
void gemm_wmma_kernel(const f16* __restrict__ A, int lda,
                      const void* __restrict__ Bsrc, int ldb,
                      const float* __restrict__ bias,
                      const float* __restrict__ residual, int ldr,
                      float* __restrict__ outF32, int ldc,
                      f16* __restrict__ outH, int ldh,
                      int M, int N, int K) {
  __shared__ f16 aT[128 * 40];   // [row][k], stride 40 to de-bank
  __shared__ f16 bT[64 * 40];    // transposed: [n][k], stride 40

  const int tid   = threadIdx.x;
  const int lane  = tid & 31;
  const int wave  = tid >> 5;
  const int wm    = wave & 3;          // 4 wave-rows of 32
  const int wn    = wave >> 2;         // 2 wave-cols of 32
  const int l16   = lane & 15;
  const int lhalf = lane >> 4;
  const int m0blk = blockIdx.y * 128;
  const int n0blk = blockIdx.x * 64;

  v8f acc[2][2] = {};

  const int nk = (K + 31) >> 5;
  for (int kt = 0; kt < nk; ++kt) {
    const int k0 = kt << 5;

    // ---- stage A tile: 128 x 32 f16
    {
      const int r  = tid >> 1;
      const int c0 = (tid & 1) << 4;
      const int gm = m0blk + r;
      f16* dst = &aT[r * 40 + c0];
      if (gm < M && (k0 + 32) <= K) {
        const f16* src = A + (size_t)gm * lda + k0 + c0;
        if (kt + 1 < nk) __builtin_prefetch(A + (size_t)gm * lda + k0 + 32, 0, 1);
        copy32_g2l(dst, src);
      } else {
        // clamped unconditional loads + select (no divergent branches)
        const int gmc = (gm < M) ? gm : (M - 1);
#pragma unroll
        for (int c = 0; c < 16; ++c) {
          const int k  = k0 + c0 + c;
          const int kc = (k < K) ? k : (K - 1);
          const f16 v = A[(size_t)gmc * lda + kc];
          dst[c] = (gm < M && k < K) ? v : (f16)0.f;
        }
      }
    }

    // ---- stage B tile into bT[n][k] (32 x 64 logical)
    if (BMODE == 0) {
      const int k  = tid >> 3;
      const int c0 = (tid & 7) << 3;
      const int gk = k0 + k;
      const float* Bf = (const float*)Bsrc;
      if (gk < K && (n0blk + 64) <= N) {
        const float4* src = (const float4*)(Bf + (size_t)gk * ldb + n0blk + c0);
        const float4 x0 = src[0], x1 = src[1];
        const float vv[8] = {x0.x, x0.y, x0.z, x0.w, x1.x, x1.y, x1.z, x1.w};
#pragma unroll
        for (int i = 0; i < 8; ++i) bT[(c0 + i) * 40 + k] = (f16)vv[i];
      } else {
        const int gkc = (gk < K) ? gk : (K - 1);
#pragma unroll
        for (int i = 0; i < 8; ++i) {
          const int n  = n0blk + c0 + i;
          const int nc = (n < N) ? n : (N - 1);
          const float v = Bf[(size_t)gkc * ldb + nc];
          bT[(c0 + i) * 40 + k] = (gk < K && n < N) ? (f16)v : (f16)0.f;
        }
      }
    } else if (BMODE == 1) {
      const int k  = tid >> 3;
      const int c0 = (tid & 7) << 3;
      const int gk = k0 + k;
      const f16* Bh = (const f16*)Bsrc;
      if (gk < K && (n0blk + 64) <= N) {
        const uint4 x = *(const uint4*)(Bh + (size_t)gk * ldb + n0blk + c0);
        const f16* xv = (const f16*)&x;
#pragma unroll
        for (int i = 0; i < 8; ++i) bT[(c0 + i) * 40 + k] = xv[i];
      } else {
        const int gkc = (gk < K) ? gk : (K - 1);
#pragma unroll
        for (int i = 0; i < 8; ++i) {
          const int n  = n0blk + c0 + i;
          const int nc = (n < N) ? n : (N - 1);
          const f16 v = Bh[(size_t)gkc * ldb + nc];
          bT[(c0 + i) * 40 + k] = (gk < K && n < N) ? v : (f16)0.f;
        }
      }
    } else {  // BMODE == 2 : B[k][n] = Bh[n*ldb + k] (contiguous along k)
      const int n   = tid >> 2;
      const int kc0 = (tid & 3) << 3;
      const int gn  = n0blk + n;
      const f16* Bh = (const f16*)Bsrc;
      f16* dst = &bT[n * 40 + kc0];
      if (gn < N && (k0 + kc0 + 8) <= K) {
        copy16_g2l(dst, Bh + (size_t)gn * ldb + k0 + kc0);
      } else {
        const int gnc = (gn < N) ? gn : (N - 1);
#pragma unroll
        for (int i = 0; i < 8; ++i) {
          const int k  = k0 + kc0 + i;
          const int kc = (k < K) ? k : (K - 1);
          const f16 v = Bh[(size_t)gnc * ldb + kc];
          dst[i] = (gn < N && k < K) ? v : (f16)0.f;
        }
      }
    }
    wait_async_lds();
    __syncthreads();

    // ---- fragment loads from LDS (ISA 16-bit A/B layouts, wave32)
    HFrag a0, a1, b0, b1;
#pragma unroll
    for (int p = 0; p < 8; ++p) {
      const int kk = ((p < 4) ? 0 : 16) + (lhalf << 3) + ((p & 3) << 1);
      const int m  = wm * 32 + l16;
      a0.u[p] = *(const unsigned int*)&aT[m * 40 + kk];
      a1.u[p] = *(const unsigned int*)&aT[(m + 16) * 40 + kk];
      const int kb = ((p < 4) ? 0 : 16) + (lhalf << 3) + ((p & 3) << 1);
      const int n  = wn * 32 + l16;
      b0.u[p] = *(const unsigned int*)&bT[n * 40 + kb];
      b1.u[p] = *(const unsigned int*)&bT[(n + 16) * 40 + kb];
    }

    acc[0][0] = __builtin_amdgcn_wmma_f32_16x16x32_f16(false, a0.v, false, b0.v,
                                                       (short)0, acc[0][0], false, false);
    acc[0][1] = __builtin_amdgcn_wmma_f32_16x16x32_f16(false, a0.v, false, b1.v,
                                                       (short)0, acc[0][1], false, false);
    acc[1][0] = __builtin_amdgcn_wmma_f32_16x16x32_f16(false, a1.v, false, b0.v,
                                                       (short)0, acc[1][0], false, false);
    acc[1][1] = __builtin_amdgcn_wmma_f32_16x16x32_f16(false, a1.v, false, b1.v,
                                                       (short)0, acc[1][1], false, false);
    __syncthreads();
  }

  // ---- epilogue: C layout row = r + 8*half, col = lane&15
#pragma unroll
  for (int i = 0; i < 2; ++i) {
#pragma unroll
    for (int j = 0; j < 2; ++j) {
      const int col = n0blk + wn * 32 + j * 16 + l16;
      if (col >= N) continue;
      const float bval = bias ? bias[col] : 0.f;
#pragma unroll
      for (int r = 0; r < 8; ++r) {
        const int row = m0blk + wm * 32 + i * 16 + (lhalf << 3) + r;
        if (row >= M) continue;
        float v = acc[i][j][r] + bval;
        if (residual) v += residual[(size_t)row * ldr + col];
        if (outF32) outF32[(size_t)row * ldc + col] = v;
        if (outH)   outH[(size_t)row * ldh + col] = (f16)v;
      }
    }
  }
}

// ---------------------------------------------------------------- LayerNorm (D=1024), f32 in -> f16 out
__global__ __launch_bounds__(TBS)
void ln_kernel(const float* __restrict__ x, const float* __restrict__ g,
               const float* __restrict__ bvec, f16* __restrict__ out) {
  __shared__ float red[8];
  const int tid = threadIdx.x;
  const size_t row = blockIdx.x;
  const float* xr = x + row * 1024;
  f16* orow = out + row * 1024;

  float v[4];
  float s = 0.f;
#pragma unroll
  for (int i = 0; i < 4; ++i) { v[i] = xr[tid + 256 * i]; s += v[i]; }
  s = block_reduce_sum(s, red);
  const float mu = s * (1.0f / 1024.0f);

  float s2 = 0.f;
#pragma unroll
  for (int i = 0; i < 4; ++i) { const float d = v[i] - mu; s2 += d * d; }
  s2 = block_reduce_sum(s2, red);
  const float rs = rsqrtf(s2 * (1.0f / 1024.0f) + 1e-5f);

#pragma unroll
  for (int i = 0; i < 4; ++i) {
    const int c = tid + 256 * i;
    orow[c] = (f16)((v[i] - mu) * rs * g[c] + bvec[c]);
  }
}

// ---------------------------------------------------------------- row softmax: P = softmax(S*scale), f16 out, zero-pad to Lp
__global__ __launch_bounds__(TBS)
void softmax_kernel(const float* __restrict__ S, int ldS, f16* __restrict__ P,
                    int ldP, int L, int Lp, float scale) {
  __shared__ float red[8];
  const int tid = threadIdx.x;
  const size_t row = blockIdx.x;
  const float* sr = S + row * (size_t)ldS;
  f16* pr = P + row * (size_t)ldP;

  float mx = -3.0e38f;
  for (int c = tid; c < L; c += TBS) mx = fmaxf(mx, sr[c] * scale);
  mx = block_reduce_max(mx, red);

  float sum = 0.f;
  for (int c = tid; c < L; c += TBS) sum += __expf(sr[c] * scale - mx);
  sum = block_reduce_sum(sum, red);
  const float inv = 1.0f / sum;

  for (int c = tid; c < Lp; c += TBS)
    pr[c] = (c < L) ? (f16)(__expf(sr[c] * scale - mx) * inv) : (f16)0.f;
}

// ---------------------------------------------------------------- GEGLU: out = val * gelu_exact(gate)
__global__ __launch_bounds__(TBS)
void geglu_kernel(const f16* __restrict__ p, f16* __restrict__ out,
                  int rows, int ff) {
  const size_t total = (size_t)rows * ff;
  const size_t stride = (size_t)gridDim.x * TBS;
  for (size_t i = (size_t)blockIdx.x * TBS + threadIdx.x; i < total; i += stride) {
    const size_t r = i / ff;
    const size_t c = i - r * ff;
    const float val  = (float)p[r * (size_t)(2 * ff) + c];
    const float gate = (float)p[r * (size_t)(2 * ff) + ff + c];
    const float ge = 0.5f * gate * (1.0f + erff(gate * 0.70710678118654752f));
    out[r * (size_t)ff + c] = (f16)(val * ge);
  }
}

// ---------------------------------------------------------------- f32 -> f16 convert
__global__ __launch_bounds__(TBS)
void f32_to_f16_kernel(const float* __restrict__ in, f16* __restrict__ out, size_t n) {
  const size_t i = (size_t)blockIdx.x * TBS + threadIdx.x;
  if (i < n) out[i] = (f16)in[i];
}

// ---------------------------------------------------------------- host
extern "C" void kernel_launch(void* const* d_in, const int* in_sizes, int n_in,
                              void* d_out, int out_size, void* d_ws, size_t ws_size,
                              hipStream_t stream) {
  (void)in_sizes; (void)n_in; (void)out_size; (void)ws_size;
  constexpr int Bz = 2, Nn = 2048, Dd = 1024, NCk = 77, DCk = 768, Hh = 16, DH = 64, FF = 4096;
  constexpr float SCALE = 0.125f;  // 64^-0.5
  const int R = Bz * Nn;           // 4096 token rows

  const float* x     = (const float*)d_in[0];
  const float* ctx   = (const float*)d_in[1];
  const float* w1_q  = (const float*)d_in[2];
  const float* w1_k  = (const float*)d_in[3];
  const float* w1_v  = (const float*)d_in[4];
  const float* w1_o  = (const float*)d_in[5];
  const float* b1_o  = (const float*)d_in[6];
  const float* w2_q  = (const float*)d_in[7];
  const float* w2_k  = (const float*)d_in[8];
  const float* w2_v  = (const float*)d_in[9];
  const float* w2_o  = (const float*)d_in[10];
  const float* b2_o  = (const float*)d_in[11];
  const float* ln1_g = (const float*)d_in[12];
  const float* ln1_b = (const float*)d_in[13];
  const float* ln2_g = (const float*)d_in[14];
  const float* ln2_b = (const float*)d_in[15];
  const float* ln3_g = (const float*)d_in[16];
  const float* ln3_b = (const float*)d_in[17];
  const float* ff_w1 = (const float*)d_in[18];
  const float* ff_b1 = (const float*)d_in[19];
  const float* ff_w2 = (const float*)d_in[20];
  const float* ff_b2 = (const float*)d_in[21];

  char* ws = (char*)d_ws;
  const size_t MB = 1ull << 20;
  f16*   h16 = (f16*)(ws);               //  8 MB  LN output (reused 3x)
  float* x1  = (float*)(ws + 8 * MB);    // 16 MB  x after self-attn
  float* x2  = (float*)(ws + 24 * MB);   // 16 MB  x after cross-attn
  char*  sc  = ws + 40 * MB;             // 96 MB  shared scratch region
  // attention phase layout of scratch
  f16*   q16  = (f16*)(sc);
  f16*   k16  = (f16*)(sc + 8 * MB);
  f16*   v16  = (f16*)(sc + 16 * MB);
  f16*   o16  = (f16*)(sc + 24 * MB);
  float* Sbuf = (float*)(sc + 32 * MB);  // 16 MB (one (b,h) score matrix)
  f16*   P16  = (f16*)(sc + 48 * MB);    //  8 MB
  f16*   c16  = (f16*)(sc + 56 * MB);    //  0.25 MB context f16
  // FF phase layout of scratch (aliases attention buffers, no longer live)
  f16*   p16  = (f16*)(sc);              // 64 MB  h @ ff_w1 + b1
  f16*   vg16 = (f16*)(sc + 64 * MB);    // 32 MB  val * gelu(gate)

  const dim3 blk(TBS);
  const dim3 gProj(Dd / 64, R / 128);        // (16, 32) for 4096x1024 outputs

  // ---- LN1 + self-attention ----
  ln_kernel<<<R, blk, 0, stream>>>(x, ln1_g, ln1_b, h16);
  gemm_wmma_kernel<0><<<gProj, blk, 0, stream>>>(h16, Dd, w1_q, Dd, nullptr, nullptr, 0,
                                                 nullptr, 0, q16, Dd, R, Dd, Dd);
  gemm_wmma_kernel<0><<<gProj, blk, 0, stream>>>(h16, Dd, w1_k, Dd, nullptr, nullptr, 0,
                                                 nullptr, 0, k16, Dd, R, Dd, Dd);
  gemm_wmma_kernel<0><<<gProj, blk, 0, stream>>>(h16, Dd, w1_v, Dd, nullptr, nullptr, 0,
                                                 nullptr, 0, v16, Dd, R, Dd, Dd);
  for (int b = 0; b < Bz; ++b) {
    for (int hh = 0; hh < Hh; ++hh) {
      const f16* qh = q16 + (size_t)b * Nn * Dd + hh * DH;
      const f16* kh = k16 + (size_t)b * Nn * Dd + hh * DH;
      const f16* vh = v16 + (size_t)b * Nn * Dd + hh * DH;
      f16*       oh = o16 + (size_t)b * Nn * Dd + hh * DH;
      gemm_wmma_kernel<2><<<dim3(Nn / 64, Nn / 128), blk, 0, stream>>>(
          qh, Dd, kh, Dd, nullptr, nullptr, 0, Sbuf, Nn, nullptr, 0, Nn, Nn, DH);
      softmax_kernel<<<Nn, blk, 0, stream>>>(Sbuf, Nn, P16, Nn, Nn, Nn, SCALE);
      gemm_wmma_kernel<1><<<dim3(1, Nn / 128), blk, 0, stream>>>(
          P16, Nn, vh, Dd, nullptr, nullptr, 0, nullptr, 0, oh, Dd, Nn, DH, Nn);
    }
  }
  gemm_wmma_kernel<0><<<gProj, blk, 0, stream>>>(o16, Dd, w1_o, Dd, b1_o, x, Dd,
                                                 x1, Dd, nullptr, 0, R, Dd, Dd);

  // ---- LN2 + cross-attention ----
  ln_kernel<<<R, blk, 0, stream>>>(x1, ln2_g, ln2_b, h16);
  {
    const size_t nctx = (size_t)Bz * NCk * DCk;
    f32_to_f16_kernel<<<(unsigned)((nctx + TBS - 1) / TBS), blk, 0, stream>>>(ctx, c16, nctx);
  }
  gemm_wmma_kernel<0><<<gProj, blk, 0, stream>>>(h16, Dd, w2_q, Dd, nullptr, nullptr, 0,
                                                 nullptr, 0, q16, Dd, R, Dd, Dd);
  const dim3 gKV(Dd / 64, (Bz * NCk + 127) / 128);  // (16, 2), M = 154
  gemm_wmma_kernel<0><<<gKV, blk, 0, stream>>>(c16, DCk, w2_k, Dd, nullptr, nullptr, 0,
                                               nullptr, 0, k16, Dd, Bz * NCk, Dd, DCk);
  gemm_wmma_kernel<0><<<gKV, blk, 0, stream>>>(c16, DCk, w2_v, Dd, nullptr, nullptr, 0,
                                               nullptr, 0, v16, Dd, Bz * NCk, Dd, DCk);
  for (int b = 0; b < Bz; ++b) {
    for (int hh = 0; hh < Hh; ++hh) {
      const f16* qh = q16 + (size_t)b * Nn * Dd + hh * DH;
      const f16* kh = k16 + (size_t)b * NCk * Dd + hh * DH;
      const f16* vh = v16 + (size_t)b * NCk * Dd + hh * DH;
      f16*       oh = o16 + (size_t)b * Nn * Dd + hh * DH;
      gemm_wmma_kernel<2><<<dim3((NCk + 63) / 64, Nn / 128), blk, 0, stream>>>(
          qh, Dd, kh, Dd, nullptr, nullptr, 0, Sbuf, 80, nullptr, 0, Nn, NCk, DH);
      softmax_kernel<<<Nn, blk, 0, stream>>>(Sbuf, 80, P16, 128, NCk, 128, SCALE);
      gemm_wmma_kernel<1><<<dim3(1, Nn / 128), blk, 0, stream>>>(
          P16, 128, vh, Dd, nullptr, nullptr, 0, nullptr, 0, oh, Dd, Nn, DH, NCk);
    }
  }
  gemm_wmma_kernel<0><<<gProj, blk, 0, stream>>>(o16, Dd, w2_o, Dd, b2_o, x1, Dd,
                                                 x2, Dd, nullptr, 0, R, Dd, Dd);

  // ---- LN3 + GEGLU feed-forward ----
  ln_kernel<<<R, blk, 0, stream>>>(x2, ln3_g, ln3_b, h16);
  gemm_wmma_kernel<0><<<dim3(2 * FF / 64, R / 128), blk, 0, stream>>>(
      h16, Dd, ff_w1, 2 * FF, ff_b1, nullptr, 0, nullptr, 0, p16, 2 * FF, R, 2 * FF, Dd);
  geglu_kernel<<<8192, blk, 0, stream>>>(p16, vg16, R, FF);
  gemm_wmma_kernel<0><<<gProj, blk, 0, stream>>>(vg16, FF, ff_w2, Dd, ff_b2, x2, Dd,
                                                 (float*)d_out, Dd, nullptr, 0, R, Dd, FF);
}
